// VGG16RoIHead_75041668595813
// MI455X (gfx1250) — matrix operand order repaired and checked by reading, not compiled, not run on hardware
//
#include <hip/hip_runtime.h>
#include <math.h>

// ---------------------------------------------------------------------------
// Types for CDNA5 WMMA (gfx1250, wave32)
// ---------------------------------------------------------------------------
typedef __attribute__((ext_vector_type(16))) __bf16 v16bf;
typedef __attribute__((ext_vector_type(8)))  float  v8f;

union FragBF16 {
    v16bf v;
    uint4 u[2];
};

static constexpr int BM = 128;
static constexpr int BN = 128;
static constexpr int BK = 32;
static constexpr int LDSW = 40;   // LDS row stride in bf16 elems (32 data + 8 pad -> conflict-free b128 reads)

// ---------------------------------------------------------------------------
// Kernel 1: Caffe-style RoI max pooling -> flat[R, C*49]
// ---------------------------------------------------------------------------
__global__ __launch_bounds__(256)
void roipool_kernel(const float* __restrict__ x, const float* __restrict__ rois,
                    const int* __restrict__ roi_idx, float* __restrict__ flat)
{
    constexpr int C = 512, H = 50, W = 50, P = 7;
    const int r = blockIdx.x;

    const float ymin = rois[r * 4 + 0];
    const float xmin = rois[r * 4 + 1];
    const float ymax = rois[r * 4 + 2];
    const float xmax = rois[r * 4 + 3];
    const int   img_idx = roi_idx[r];

    // jnp.round == round-half-to-even == rintf (default RNE)
    const float sw = rintf(xmin * 0.0625f);
    const float sh = rintf(ymin * 0.0625f);
    const float ew = rintf(xmax * 0.0625f);
    const float eh = rintf(ymax * 0.0625f);
    const float bw = fmaxf(ew - sw + 1.0f, 1.0f) * (1.0f / 7.0f);
    const float bh = fmaxf(eh - sh + 1.0f, 1.0f) * (1.0f / 7.0f);

    int wsA[P], weA[P], hsA[P], heA[P];
#pragma unroll
    for (int p = 0; p < P; ++p) {
        wsA[p] = (int)fminf(fmaxf(floorf((float)p * bw) + sw, 0.0f), (float)W);
        weA[p] = (int)fminf(fmaxf(ceilf(((float)p + 1.0f) * bw) + sw, 0.0f), (float)W);
        hsA[p] = (int)fminf(fmaxf(floorf((float)p * bh) + sh, 0.0f), (float)H);
        heA[p] = (int)fminf(fmaxf(ceilf(((float)p + 1.0f) * bh) + sh, 0.0f), (float)H);
    }

    for (int c = threadIdx.x; c < C; c += blockDim.x) {
        const float* img = x + ((size_t)img_idx * C + c) * (size_t)(H * W);
        float* out = flat + (size_t)r * (C * P * P) + (size_t)c * (P * P);
#pragma unroll 1
        for (int ph = 0; ph < P; ++ph) {
#pragma unroll 1
            for (int pw = 0; pw < P; ++pw) {
                float m = -INFINITY;
                for (int h = hsA[ph]; h < heA[ph]; ++h)
                    for (int w = wsA[pw]; w < weA[pw]; ++w)
                        m = fmaxf(m, img[h * W + w]);
                // empty window -> -inf -> 0 (reference: where(isfinite, m, 0))
                out[ph * P + pw] = (m == -INFINITY) ? 0.0f : m;
            }
        }
    }
}

// ---------------------------------------------------------------------------
// Kernel 2: C = act(A[MxK] @ B[KxN] + bias), fp32 in/out, split-bf16 WMMA core.
//   a = a_hi + a_lo (bf16); product = hi*hi + hi*lo + lo*hi, fp32 accumulate
//   (error ~2^-17 relative, near-fp32).
//   Double-buffered LDS + register staging: one barrier per k-step; next
//   tile's global loads are issued before the WMMA phase so HBM latency is
//   covered by matrix math.
//   Requires M % 128 == 0 and K % 32 == 0 (true for all calls); N is guarded.
// ---------------------------------------------------------------------------
__global__ __launch_bounds__(256)
void gemm_bf16split(const float* __restrict__ A, const float* __restrict__ B,
                    const float* __restrict__ bias, float* __restrict__ C,
                    int M, int N, int K, int relu)
{
    __shared__ __align__(16) __bf16 sAhi[2][BM * LDSW];
    __shared__ __align__(16) __bf16 sAlo[2][BM * LDSW];
    __shared__ __align__(16) __bf16 sBhi[2][BN * LDSW];  // stored transposed: [n][k]
    __shared__ __align__(16) __bf16 sBlo[2][BN * LDSW];

    const int tid  = threadIdx.x;
    const int lane = tid & 31;
    const int wave = tid >> 5;
    const int wm   = wave >> 2;     // 0..1 : 64-row slab
    const int wn   = wave & 3;      // 0..3 : 32-col slab
    const int l15  = lane & 15;
    const int lh   = lane >> 4;     // lane half selects 16-wide K block (k = 16*lh + s)

    const int m0 = blockIdx.x * BM;
    const int n0 = blockIdx.y * BN;

    // ---- register staging for the next global tile ----
    float4 aReg[4];
    float  bReg[4][4];

    auto loadGlobal = [&](int kt) {
#pragma unroll
        for (int p = 0; p < 4; ++p) {
            int lin = p * 256 + tid;          // 0..1023
            int row = lin >> 3;               // 0..127
            int c4  = (lin & 7) << 2;         // 0..28
            aReg[p] = *(const float4*)(A + (size_t)(m0 + row) * K + kt + c4);
        }
#pragma unroll
        for (int p = 0; p < 4; ++p) {
            int lin = p * 256 + tid;
            int kr  = lin >> 5;               // 0..31
            int n4  = (lin & 31) << 2;        // 0..124
#pragma unroll
            for (int j = 0; j < 4; ++j) {
                int n = n0 + n4 + j;
                bReg[p][j] = (n < N) ? B[(size_t)(kt + kr) * N + n] : 0.0f;
            }
        }
    };

    auto storeLds = [&](int buf) {
#pragma unroll
        for (int p = 0; p < 4; ++p) {
            int lin = p * 256 + tid;
            int row = lin >> 3;
            int c4  = (lin & 7) << 2;
            const float* af = (const float*)&aReg[p];
#pragma unroll
            for (int j = 0; j < 4; ++j) {
                float  f = af[j];
                __bf16 h = (__bf16)f;
                __bf16 l = (__bf16)(f - (float)h);
                sAhi[buf][row * LDSW + c4 + j] = h;
                sAlo[buf][row * LDSW + c4 + j] = l;
            }
        }
#pragma unroll
        for (int p = 0; p < 4; ++p) {
            int lin = p * 256 + tid;
            int kr  = lin >> 5;
            int n4  = (lin & 31) << 2;
#pragma unroll
            for (int j = 0; j < 4; ++j) {
                float  f = bReg[p][j];
                __bf16 h = (__bf16)f;
                __bf16 l = (__bf16)(f - (float)h);
                sBhi[buf][(n4 + j) * LDSW + kr] = h;
                sBlo[buf][(n4 + j) * LDSW + kr] = l;
            }
        }
    };

    v8f acc[4][2];
#pragma unroll
    for (int i = 0; i < 4; ++i)
#pragma unroll
        for (int j = 0; j < 2; ++j)
#pragma unroll
            for (int e = 0; e < 8; ++e) acc[i][j][e] = 0.0f;

    loadGlobal(0);
    int buf = 0;

    for (int kt = 0; kt < K; kt += BK) {
        storeLds(buf);
        __syncthreads();                       // LDS[buf] tile complete

        if (kt + BK < K) loadGlobal(kt + BK);  // overlap HBM with WMMA below

        // ---- B fragments: held across the k-step (contiguous k = 16*lh + s, as for A) ----
        FragBF16 fbhi[2], fblo[2];
#pragma unroll
        for (int jn = 0; jn < 2; ++jn) {
            int col = wn * 32 + jn * 16 + l15;
            const __bf16* ph = &sBhi[buf][col * LDSW + lh * 16];
            const __bf16* pl = &sBlo[buf][col * LDSW + lh * 16];
            fbhi[jn].u[0] = *(const uint4*)(ph);        // K = lh*16 + 0..7
            fbhi[jn].u[1] = *(const uint4*)(ph + 8);    // K = lh*16 + 8..15
            fblo[jn].u[0] = *(const uint4*)(pl);
            fblo[jn].u[1] = *(const uint4*)(pl + 8);
        }

        // ---- stream A fragments; 6 WMMAs per m-subtile ----
#pragma unroll
        for (int im = 0; im < 4; ++im) {
            int row = wm * 64 + im * 16 + l15;
            const __bf16* ph = &sAhi[buf][row * LDSW + lh * 16];
            const __bf16* pl = &sAlo[buf][row * LDSW + lh * 16];
            FragBF16 fahi, falo;
            fahi.u[0] = *(const uint4*)(ph);            // K = lh*16 + 0..7
            fahi.u[1] = *(const uint4*)(ph + 8);        // K = lh*16 + 8..15
            falo.u[0] = *(const uint4*)(pl);
            falo.u[1] = *(const uint4*)(pl + 8);
#pragma unroll
            for (int jn = 0; jn < 2; ++jn) {
                acc[im][jn] = __builtin_amdgcn_wmma_f32_16x16x32_bf16(
                    false, fahi.v, false, fbhi[jn].v, (short)0, acc[im][jn], false, false);
                acc[im][jn] = __builtin_amdgcn_wmma_f32_16x16x32_bf16(
                    false, fahi.v, false, fblo[jn].v, (short)0, acc[im][jn], false, false);
                acc[im][jn] = __builtin_amdgcn_wmma_f32_16x16x32_bf16(
                    false, falo.v, false, fbhi[jn].v, (short)0, acc[im][jn], false, false);
            }
        }
        buf ^= 1;
    }

    // ---- epilogue: C layout = VGPR r -> M=r (lanes 0-15), M=r+8 (lanes 16-31) ----
#pragma unroll
    for (int im = 0; im < 4; ++im) {
#pragma unroll
        for (int jn = 0; jn < 2; ++jn) {
            int col = n0 + wn * 32 + jn * 16 + l15;
            if (col >= N) continue;
            float bv = bias[col];
#pragma unroll
            for (int rr = 0; rr < 8; ++rr) {
                int row = m0 + wm * 64 + im * 16 + rr + 8 * lh;
                float v = acc[im][jn][rr] + bv;
                if (relu) v = fmaxf(v, 0.0f);
                C[(size_t)row * N + col] = v;
            }
        }
    }
}

// ---------------------------------------------------------------------------
// Launch
// ---------------------------------------------------------------------------
extern "C" void kernel_launch(void* const* d_in, const int* in_sizes, int n_in,
                              void* d_out, int out_size, void* d_ws, size_t ws_size,
                              hipStream_t stream)
{
    const float* x    = (const float*)d_in[0];   // [4,512,50,50]
    const float* rois = (const float*)d_in[1];   // [512,4]
    const int*   ridx = (const int*)d_in[2];     // [512]
    const float* W1   = (const float*)d_in[3];   // [25088,4096]
    const float* b1   = (const float*)d_in[4];   // [4096]
    const float* W2   = (const float*)d_in[5];   // [4096,4096]
    const float* b2   = (const float*)d_in[6];   // [4096]
    const float* Wc   = (const float*)d_in[7];   // [4096,84]
    const float* bc   = (const float*)d_in[8];   // [84]
    const float* Ws   = (const float*)d_in[9];   // [4096,21]
    const float* bs   = (const float*)d_in[10];  // [21]
    float* out = (float*)d_out;

    const int R = 512, DIN = 512 * 49 /*25088*/, HID = 4096, NLOC = 84, NSC = 21;

    // Workspace: flat [512,25088] | fc6 [512,4096] | fc7 [512,4096]  (~68 MB)
    float* FLAT = (float*)d_ws;
    float* FC6  = FLAT + (size_t)R * DIN;
    float* FC7  = FC6 + (size_t)R * HID;

    roipool_kernel<<<R, 256, 0, stream>>>(x, rois, ridx, FLAT);

    // fc6 = relu(flat @ W1 + b1)
    gemm_bf16split<<<dim3(R / BM, HID / BN), 256, 0, stream>>>(FLAT, W1, b1, FC6, R, HID, DIN, 1);
    // fc7 = relu(fc6 @ W2 + b2)
    gemm_bf16split<<<dim3(R / BM, HID / BN), 256, 0, stream>>>(FC6, W2, b2, FC7, R, HID, HID, 1);
    // roi_cls_locs = fc7 @ Wc + bc
    gemm_bf16split<<<dim3(R / BM, 1), 256, 0, stream>>>(FC7, Wc, bc, out, R, NLOC, HID, 0);
    // roi_scores = fc7 @ Ws + bs
    gemm_bf16split<<<dim3(R / BM, 1), 256, 0, stream>>>(FC7, Ws, bs, out + (size_t)R * NLOC, R, NSC, HID, 0);
}
